// MPNN_18889266168017
// MI455X (gfx1250) — matrix operand order repaired and checked by reading, not compiled, not run on hardware
//
#include <hip/hip_runtime.h>
#include <hip/hip_bf16.h>

// ---------------------------------------------------------------------------
// MPNN (2x GCNConv + BN + ReLU, MLP head) for MI455X (gfx1250, wave32).
// GEMMs use v_wmma_f32_16x16x32_bf16 (bf16 A/B, f32 accum).
// Edge scatter uses global_atomic_add_f32; working set is L2-resident (192MB).
// ---------------------------------------------------------------------------

typedef __attribute__((ext_vector_type(16))) __bf16 v16bf;
typedef __attribute__((ext_vector_type(8)))  float  v8f;

#define NF 128          // feature width (NFEAT == NHID == 128)
#define TM 128          // GEMM rows per workgroup
#define PAD 4           // LDS row padding (bf16 elems) to spread banks

// ---------------------------------------------------------------- utilities
__global__ __launch_bounds__(256) void k_zero(float* __restrict__ p, long long n) {
    long long i = (long long)blockIdx.x * 256 + threadIdx.x;
    long long st = (long long)gridDim.x * 256;
    for (; i < n; i += st) p[i] = 0.0f;
}

// ------------------------------------------------------------- degree / dinv
__global__ __launch_bounds__(256) void k_degree(const int* __restrict__ ei,
                                                const float* __restrict__ ew,
                                                float* __restrict__ deg, int E_) {
    int e = blockIdx.x * 256 + threadIdx.x;
    if (e >= E_) return;
    int dst = ei[E_ + e];
    unsafeAtomicAdd(&deg[dst], ew[e]);
}

__global__ __launch_bounds__(256) void k_dinv(const float* __restrict__ deg,
                                              float* __restrict__ dinv, int n) {
    int i = blockIdx.x * 256 + threadIdx.x;
    if (i >= n) return;
    dinv[i] = rsqrtf(deg[i] + 1.0f);   // +1 self-loop weight
}

// ------------------------------------------------------------- WMMA GEMM
// Out[M x 128] = relu?( A[M x (128*kchunks)] @ W[(128*kchunks) x 128] + bias )
// A is split in up-to-3 parts of 128 cols each (for the concat [x|h1|h2]).
// One workgroup: 256 threads (8 waves) computes a TM x 128 tile.
// Wave w owns rows [w*16, w*16+16); loops 8 N-tiles reusing its A fragment.
__global__ __launch_bounds__(256) void k_gemm_wmma(
    const float* __restrict__ A0, const float* __restrict__ A1,
    const float* __restrict__ A2, const float* __restrict__ W,
    const float* __restrict__ bias, float* __restrict__ Out,
    int M, int kchunks, int do_relu)
{
    __shared__ __bf16 sA[TM][NF + PAD];        // A chunk (bf16), row-major
    __shared__ __bf16 sWt[NF][NF + PAD];       // W chunk transposed: [n][k]

    const int tid  = threadIdx.x;
    const int lane = tid & 31;
    const int wave = tid >> 5;                 // 0..7
    const int half = lane >> 4;                // 0/1
    const int l16  = lane & 15;
    const int rowBase = blockIdx.x * TM;

    v8f acc[8];
    #pragma unroll
    for (int nt = 0; nt < 8; ++nt) acc[nt] = (v8f){0,0,0,0,0,0,0,0};

    for (int kc = 0; kc < kchunks; ++kc) {
        const float* Asrc = (kc == 0) ? A0 : ((kc == 1) ? A1 : A2);

        // ---- cooperative load: A chunk (fp32 -> bf16), 16 float4 per thread
        #pragma unroll 4
        for (int i = 0; i < 16; ++i) {
            int linear = tid + i * 256;        // 0..4095 float4 slots
            int row = linear >> 5;             // 32 float4 per row
            int c4  = linear & 31;
            float4 v = make_float4(0.f, 0.f, 0.f, 0.f);
            int grow = rowBase + row;
            if (grow < M)
                v = *(const float4*)(Asrc + (long long)grow * NF + c4 * 4);
            sA[row][c4 * 4 + 0] = (__bf16)v.x;
            sA[row][c4 * 4 + 1] = (__bf16)v.y;
            sA[row][c4 * 4 + 2] = (__bf16)v.z;
            sA[row][c4 * 4 + 3] = (__bf16)v.w;
        }
        // ---- cooperative load: W chunk transposed into LDS
        #pragma unroll 4
        for (int i = 0; i < 16; ++i) {
            int linear = tid + i * 256;
            int k  = linear >> 5;
            int c4 = linear & 31;
            float4 v = *(const float4*)(W + (long long)(kc * NF + k) * NF + c4 * 4);
            sWt[c4 * 4 + 0][k] = (__bf16)v.x;
            sWt[c4 * 4 + 1][k] = (__bf16)v.y;
            sWt[c4 * 4 + 2][k] = (__bf16)v.z;
            sWt[c4 * 4 + 3][k] = (__bf16)v.w;
        }
        __syncthreads();

        // ---- 4 K-steps of 32, 8 N-tiles each
        #pragma unroll
        for (int kt = 0; kt < 4; ++kt) {
            // A fragment: 16-bit A 16x32 layout (lane half selects K sub-block)
            v16bf afrag;
            const __bf16* ap = &sA[wave * 16 + l16][kt * 32 + half * 16];
            #pragma unroll
            for (int e = 0; e < 8; ++e) afrag[e]     = ap[e];
            #pragma unroll
            for (int e = 0; e < 8; ++e) afrag[8 + e] = ap[8 + e];

            #pragma unroll
            for (int nt = 0; nt < 8; ++nt) {
                // B fragment: B 32x16, lane = col, contiguous K in sWt row
                v16bf bfrag;
                const __bf16* bp = &sWt[nt * 16 + l16][kt * 32 + half * 16];
                #pragma unroll
                for (int e = 0; e < 16; ++e) bfrag[e] = bp[e];

                acc[nt] = __builtin_amdgcn_wmma_f32_16x16x32_bf16(
                    false, afrag, false, bfrag, (short)0, acc[nt], false, false);
            }
        }
        __syncthreads();
    }

    // ---- epilogue: bias (+relu), store.  D vgpr r -> row (half*8 + r)
    #pragma unroll
    for (int nt = 0; nt < 8; ++nt) {
        int col = nt * 16 + l16;
        float bv = bias ? bias[col] : 0.0f;
        #pragma unroll
        for (int r = 0; r < 8; ++r) {
            int m = rowBase + wave * 16 + half * 8 + r;
            if (m < M) {
                float v = acc[nt][r] + bv;
                if (do_relu) v = fmaxf(v, 0.0f);
                Out[(long long)m * NF + col] = v;
            }
        }
    }
}

// ------------------------------------------------------------- edge scatter
// agg[dst] += h[src] * (dinv[src]*w*dinv[dst]);  one wave per edge, float4/lane
__global__ __launch_bounds__(256) void k_scatter(const int* __restrict__ ei,
                                                 const float* __restrict__ ew,
                                                 const float* __restrict__ dinv,
                                                 const float* __restrict__ h,
                                                 float* __restrict__ agg, int E_) {
    int gid = blockIdx.x * 256 + threadIdx.x;
    int e = gid >> 5;
    int q = gid & 31;
    if (e >= E_) return;
    int src = ei[e];
    int dst = ei[E_ + e];
    float norm = dinv[src] * ew[e] * dinv[dst];
    float4 hv = *(const float4*)(h + (long long)src * NF + q * 4);
    float* out = agg + (long long)dst * NF + q * 4;
    unsafeAtomicAdd(out + 0, hv.x * norm);
    unsafeAtomicAdd(out + 1, hv.y * norm);
    unsafeAtomicAdd(out + 2, hv.z * norm);
    unsafeAtomicAdd(out + 3, hv.w * norm);
}

// ----------------------------------- self-loop + bias + ReLU + BN partial sums
// y = relu(agg + h*dinv^2 + b)  (in place over agg); accumulate sum / sumsq.
__global__ __launch_bounds__(256) void k_finalize(const float* __restrict__ h,
                                                  float* __restrict__ agg,
                                                  const float* __restrict__ dinv,
                                                  const float* __restrict__ bias,
                                                  float* __restrict__ sums,  // [0,128)=sum [128,256)=sumsq
                                                  int n) {
    int f  = threadIdx.x & 127;
    int rg = threadIdx.x >> 7;                 // 0/1: two rows per block step
    float s = 0.f, sq = 0.f;
    float bv = bias[f];
    for (int row = blockIdx.x * 2 + rg; row < n; row += gridDim.x * 2) {
        float di = dinv[row];
        long long idx = (long long)row * NF + f;
        float y = agg[idx] + h[idx] * di * di + bv;
        y = fmaxf(y, 0.0f);
        agg[idx] = y;
        s  += y;
        sq += y * y;
    }
    __shared__ float red[2][2][128];
    red[0][rg][f] = s;
    red[1][rg][f] = sq;
    __syncthreads();
    if (rg == 0) {
        unsafeAtomicAdd(&sums[f],       red[0][0][f] + red[0][1][f]);
        unsafeAtomicAdd(&sums[128 + f], red[1][0][f] + red[1][1][f]);
    }
}

// ------------------------------------------------------------- BN scale/shift
__global__ __launch_bounds__(128) void k_bn_scale(float* __restrict__ sums,
                                                  const float* __restrict__ g,
                                                  const float* __restrict__ be,
                                                  int n) {
    int f = threadIdx.x;
    float inv_n = 1.0f / (float)n;
    float mean = sums[f] * inv_n;
    float var  = sums[128 + f] * inv_n - mean * mean;   // biased var
    float sc   = g[f] * rsqrtf(var + 1e-5f);
    sums[256 + f] = sc;
    sums[384 + f] = be[f] - mean * sc;
}

__global__ __launch_bounds__(256) void k_bn_apply(const float* __restrict__ y,
                                                  const float* __restrict__ sums,
                                                  float* __restrict__ out,
                                                  long long total) {
    long long i = (long long)blockIdx.x * 256 + threadIdx.x;
    if (i >= total) return;
    int f = (int)(i & 127);
    out[i] = y[i] * sums[256 + f] + sums[384 + f];
}

// ------------------------------------------------------------- fc2 head
// out[node] = relu( dot(mlp[node, 0:128], Wf2) + bf2 ); one wave per node.
__global__ __launch_bounds__(256) void k_head(const float* __restrict__ mlp,
                                              const float* __restrict__ Wf2,
                                              const float* __restrict__ bf2,
                                              float* __restrict__ out, int n) {
    int gid  = blockIdx.x * 256 + threadIdx.x;
    int node = gid >> 5;
    int lane = gid & 31;
    if (node >= n) return;
    float4 a = *(const float4*)(mlp + (long long)node * NF + lane * 4);
    float4 w = *(const float4*)(Wf2 + lane * 4);
    float s = a.x * w.x + a.y * w.y + a.z * w.z + a.w * w.w;
    #pragma unroll
    for (int off = 16; off > 0; off >>= 1) s += __shfl_xor(s, off, 32);
    if (lane == 0) out[node] = fmaxf(s + bf2[0], 0.0f);
}

// ---------------------------------------------------------------------------
extern "C" void kernel_launch(void* const* d_in, const int* in_sizes, int n_in,
                              void* d_out, int out_size, void* d_ws, size_t ws_size,
                              hipStream_t stream) {
    const float* x    = (const float*)d_in[0];
    const int*   ei   = (const int*)  d_in[1];
    const float* ew   = (const float*)d_in[2];
    const float* W1   = (const float*)d_in[3];
    const float* b1   = (const float*)d_in[4];
    const float* W2   = (const float*)d_in[5];
    const float* b2   = (const float*)d_in[6];
    const float* g1   = (const float*)d_in[7];
    const float* be1  = (const float*)d_in[8];
    const float* g2   = (const float*)d_in[9];
    const float* be2  = (const float*)d_in[10];
    const float* Wf1  = (const float*)d_in[11];
    const float* bf1  = (const float*)d_in[12];
    const float* Wf2  = (const float*)d_in[13];
    const float* bf2  = (const float*)d_in[14];

    const int N = in_sizes[0] / NF;
    const int E = in_sizes[2];
    float* out = (float*)d_out;

    // workspace layout (floats)
    float* ws   = (float*)d_ws;
    float* deg  = ws;                                   // N
    float* dinv = deg  + N;                             // N
    float* h    = dinv + N;                             // N*128
    float* agg  = h    + (long long)N * NF;             // N*128 (also y in place)
    float* h1   = agg  + (long long)N * NF;             // N*128
    float* h2   = h1   + (long long)N * NF;             // N*128
    float* sums = h2   + (long long)N * NF;             // 512

    const long long NFtot = (long long)N * NF;
    const int gb_feat  = (int)((NFtot + 255) / 256);            // elementwise grids
    const int gb_gemm  = (N + TM - 1) / TM;
    const int gb_edges = (E * 32 + 255) / 256;
    const int gb_head  = (N * 32 + 255) / 256;
    const int gb_zeroF = 4096;

    // ---------------- stage 0: degree / dinv (shared by both conv layers)
    k_zero<<<256, 256, 0, stream>>>(deg, N);
    k_degree<<<(E + 255) / 256, 256, 0, stream>>>(ei, ew, deg, E);
    k_dinv<<<(N + 255) / 256, 256, 0, stream>>>(deg, dinv, N);

    // ---------------- layer 1: h = x@W1 ; scatter ; relu+BN
    k_zero<<<gb_zeroF, 256, 0, stream>>>(agg, NFtot);
    k_zero<<<1, 256, 0, stream>>>(sums, 512);
    k_gemm_wmma<<<gb_gemm, 256, 0, stream>>>(x, x, x, W1, nullptr, h, N, 1, 0);
    k_scatter<<<gb_edges, 256, 0, stream>>>(ei, ew, dinv, h, agg, E);
    k_finalize<<<1024, 256, 0, stream>>>(h, agg, dinv, b1, sums, N);
    k_bn_scale<<<1, 128, 0, stream>>>(sums, g1, be1, N);
    k_bn_apply<<<gb_feat, 256, 0, stream>>>(agg, sums, h1, NFtot);

    // ---------------- layer 2: h = h1@W2 ; scatter ; relu+BN
    k_zero<<<gb_zeroF, 256, 0, stream>>>(agg, NFtot);
    k_zero<<<1, 256, 0, stream>>>(sums, 512);
    k_gemm_wmma<<<gb_gemm, 256, 0, stream>>>(h1, h1, h1, W2, nullptr, h, N, 1, 0);
    k_scatter<<<gb_edges, 256, 0, stream>>>(ei, ew, dinv, h, agg, E);
    k_finalize<<<1024, 256, 0, stream>>>(h, agg, dinv, b2, sums, N);
    k_bn_scale<<<1, 128, 0, stream>>>(sums, g2, be2, N);
    k_bn_apply<<<gb_feat, 256, 0, stream>>>(agg, sums, h2, NFtot);

    // ---------------- MLP head: relu([x|h1|h2] @ Wf1 + bf1) ; fc2 + relu
    k_gemm_wmma<<<gb_gemm, 256, 0, stream>>>(x, h1, h2, Wf1, bf1, h, N, 3, 1);
    k_head<<<gb_head, 256, 0, stream>>>(h, Wf2, bf2, out, N);
}
